// EncoderLayer_42889543418188
// MI455X (gfx1250) — compile-verified, matrix operand order checked by reading, not run on hardware
//
#include <hip/hip_runtime.h>
#include <hip/hip_bf16.h>

// ---------------------------------------------------------------------------
// MI455X (gfx1250) implementation of the spatio-temporal GAT encoder layer.
// Strategy (HBM-bound, ~12 GFLOP over ~350MB of traffic @ 23.3 TB/s):
//   * every GEMM (x@W_d/m/s, xt@Wg, concat@Wf, cov0@cov0^T) goes through
//     v_wmma_f32_16x16x32_f16 with LDS-staged f16 tiles
//   * LDS staging uses GLOBAL_LOAD_ASYNC_TO_LDS_B128 (ASYNCcnt path) when the
//     toolchain exposes it, with a synchronous b128 copy fallback
//   * dst = arange(E)%N  => each node has exactly 8 in-edges {n+i*N}, so
//     the segment softmax is a register 8-way reduction (no atomics)
//   * reference reshape quirks replicated: k_read = b*T+t, idx = kk % B
// ---------------------------------------------------------------------------

#define Bb 4
#define Tt 24
#define Nn 1024
#define Ff 96
#define Kk 96          // T*B
#define Ee 8192
#define DEG 8          // E/N edges per destination node
#define Mrows 98304    // N*K == B*T*N

typedef __attribute__((ext_vector_type(16))) _Float16 v16h;
typedef __attribute__((ext_vector_type(8)))  float    v8f;

#if defined(__has_builtin)
#if __has_builtin(__builtin_amdgcn_global_load_async_to_lds_b128) && \
    __has_builtin(__builtin_amdgcn_s_wait_asynccnt)
#define USE_ASYNC_LDS 1
#endif
#endif
#ifndef USE_ASYNC_LDS
#define USE_ASYNC_LDS 0
#endif

// payload type expected by the async-to-LDS builtin: int __vector(4) (16B)
typedef int async_b128_t __attribute__((vector_size(16)));

// 16B global -> LDS copy, async (ASYNCcnt) when available.
__device__ __forceinline__ void stage16(const uint4* __restrict__ g,
                                        uint4* __restrict__ l) {
#if USE_ASYNC_LDS
  __builtin_amdgcn_global_load_async_to_lds_b128(
      (__attribute__((address_space(1))) async_b128_t*)(void*)g,
      (__attribute__((address_space(3))) async_b128_t*)(void*)l, 0, 0);
#else
  *l = *g;
#endif
}

__device__ __forceinline__ void stage_wait() {
#if USE_ASYNC_LDS
  __builtin_amdgcn_s_wait_asynccnt(0);
#endif
}

__device__ __forceinline__ int frag_k_idx(int j, int hi) {
  // 16-bit A-matrix 16x32 layout (ISA 7.12.2): lanes 0-15 hold K={0..7,16..23},
  // lanes 16-31 hold K={8..15,24..31}; j = half index within the v16h fragment.
  int base = (j < 8) ? 0 : 16;
  return base + hi * 8 + (j & 7);
}

// ---------------------------------------------------------------------------
// Generic row-major WMMA GEMM: C(M x 96) = A(M x KD) * Bw(KD x 96)
// block = 192 threads = 6 waves; wave w owns the 16x16 tile at column w*16.
// ---------------------------------------------------------------------------
template <int KD>
__global__ __launch_bounds__(192) void wmma_gemm_rm(
    const _Float16* __restrict__ A, const _Float16* __restrict__ Bw,
    _Float16* __restrict__ C16, float* __restrict__ C32) {
  __shared__ __align__(16) _Float16 sB[KD * 96];
  __shared__ __align__(16) _Float16 sA[16 * KD];
  const int tid  = threadIdx.x;
  const int lane = tid & 31;
  const int wave = tid >> 5;                 // 0..5
  const int row0 = blockIdx.x * 16;

  if (tid == 0)  // speculative prefetch of the next row strip (global_prefetch_b8)
    __builtin_prefetch(A + ((size_t)row0 + 16) * KD, 0, 1);

  // cooperative 16B staging into LDS (async-to-LDS when available)
  {
    const uint4* gB = (const uint4*)Bw;
    uint4* lB = (uint4*)sB;
    for (int i = tid; i < (KD * 96) / 8; i += 192) stage16(gB + i, lB + i);
    const uint4* gA = (const uint4*)(A + (size_t)row0 * KD);
    uint4* lA = (uint4*)sA;
    for (int i = tid; i < (16 * KD) / 8; i += 192) stage16(gA + i, lA + i);
    stage_wait();
  }
  __syncthreads();

  const int n0 = wave * 16;
  const int m  = lane & 15;
  const int hi = lane >> 4;
  v8f c = {};
#pragma unroll
  for (int kt = 0; kt < KD / 32; ++kt) {
    v16h a, b;
#pragma unroll
    for (int j = 0; j < 16; ++j) {
      const int k = kt * 32 + frag_k_idx(j, hi);
      a[j] = sA[m * KD + k];
      b[j] = sB[k * 96 + n0 + m];
    }
    c = __builtin_amdgcn_wmma_f32_16x16x32_f16(false, a, false, b, (short)0, c,
                                               false, false);
  }
  // C/D layout: VGPR r -> row (r + 8*hi), col = lane&15
#pragma unroll
  for (int r = 0; r < 8; ++r) {
    const size_t row = (size_t)row0 + r + 8 * hi;
    const int col = n0 + m;
    if (C32) C32[row * 96 + col] = c[r];
    if (C16) C16[row * 96 + col] = (_Float16)c[r];
  }
}

// ---------------------------------------------------------------------------
// attn_S logits: per batch, cov0(1024x96) @ cov0^T -> (1024x1024) f32
// block = 256 threads = 8 waves covering a 16x128 tile.
// ---------------------------------------------------------------------------
__global__ __launch_bounds__(256) void wmma_gemm_nt_covS(
    const _Float16* __restrict__ cov, float* __restrict__ out) {
  __shared__ __align__(16) _Float16 sA[16 * 96];
  __shared__ __align__(16) _Float16 sC[128 * 96];
  const int tid = threadIdx.x, lane = tid & 31, wave = tid >> 5;
  const int b = blockIdx.z;
  const int row0 = blockIdx.x * 16;
  const int col0 = blockIdx.y * 128;
  const _Float16* covb = cov + (size_t)b * Nn * 96;
  {
    const uint4* gA = (const uint4*)(covb + (size_t)row0 * 96);
    uint4* lA = (uint4*)sA;
    for (int i = tid; i < (16 * 96) / 8; i += 256) stage16(gA + i, lA + i);
    const uint4* gC = (const uint4*)(covb + (size_t)col0 * 96);
    uint4* lC = (uint4*)sC;
    for (int i = tid; i < (128 * 96) / 8; i += 256) stage16(gC + i, lC + i);
    stage_wait();
  }
  __syncthreads();
  const int m = lane & 15, hi = lane >> 4;
  const int nloc = wave * 16 + m;
  v8f c = {};
#pragma unroll
  for (int kt = 0; kt < 3; ++kt) {
    v16h a, bb;
#pragma unroll
    for (int j = 0; j < 16; ++j) {
      const int k = kt * 32 + frag_k_idx(j, hi);
      a[j]  = sA[m * 96 + k];
      bb[j] = sC[nloc * 96 + k];   // B^T fragment: B[k][n] = cov[col0+n][k]
    }
    c = __builtin_amdgcn_wmma_f32_16x16x32_f16(false, a, false, bb, (short)0, c,
                                               false, false);
  }
#pragma unroll
  for (int r = 0; r < 8; ++r) {
    const size_t row = (size_t)row0 + r + 8 * hi;
    const size_t col = (size_t)col0 + wave * 16 + m;
    out[((size_t)b * Nn + row) * Nn + col] = c[r];
  }
}

// ---------------------------------------------------------------------------
// elementwise / staging kernels
// ---------------------------------------------------------------------------
__global__ void prep_kernel(const float* __restrict__ input,
                            const float* __restrict__ spat,
                            _Float16* __restrict__ x16,
                            _Float16* __restrict__ xt16) {
  size_t idx = (size_t)blockIdx.x * blockDim.x + threadIdx.x;
  if (idx >= (size_t)Bb * Tt * Nn * Ff) return;
  const int f = idx % Ff;
  const int n = (idx / Ff) % Nn;
  const int t = (idx / ((size_t)Ff * Nn)) % Tt;
  const int b = idx / ((size_t)Ff * Nn * Tt);
  const float vin = input[idx];
  const int k = t * Bb + b;  // x = transpose(2,1,0,3).reshape(N, T*B, F)
  x16[((size_t)n * Kk + k) * Ff + f]                    = (_Float16)(vin + spat[idx]);
  xt16[(((size_t)(b * Nn + n)) * Tt + t) * Ff + f]      = (_Float16)vin;
}

__global__ void cvt_f16_kernel(const float* __restrict__ s,
                               _Float16* __restrict__ d, int n) {
  int i = blockIdx.x * blockDim.x + threadIdx.x;
  if (i < n) d[i] = (_Float16)s[i];
}

__global__ void cov0_cvt_kernel(const float* __restrict__ spat,
                                _Float16* __restrict__ cov0) {
  int i = blockIdx.x * blockDim.x + threadIdx.x;
  if (i >= Bb * Nn * Ff) return;
  const int f = i % Ff, n = (i / Ff) % Nn, b = i / (Ff * Nn);
  cov0[i] = (_Float16)spat[(((size_t)b * Tt + 0) * Nn + n) * Ff + f];
}

// el/er reductions: el[row,h] = sum_d h[row, h*HD+d] * a[h*HD+d]
__global__ void el_er_kernel(const _Float16* __restrict__ h16,
                             const float* __restrict__ al,
                             const float* __restrict__ ar,
                             float* __restrict__ el, float* __restrict__ er,
                             int rows, int H, int HD) {
  int i = blockIdx.x * blockDim.x + threadIdx.x;
  if (i >= rows * H) return;
  const int row = i / H, h = i % H;
  const _Float16* hp = h16 + (size_t)row * 96 + h * HD;
  const float* alp = al + h * HD;
  const float* arp = ar + h * HD;
  float sl = 0.f, sr = 0.f;
  for (int d = 0; d < HD; ++d) {
    const float v = (float)hp[d];
    sl += v * alp[d];
    sr += v * arp[d];
  }
  el[i] = sl;
  er[i] = sr;
}

// GAT aggregation: one wave per (n,k); lanes = HD=32; 8 fixed edges per node.
__global__ __launch_bounds__(256) void gat_aggregate_kernel(
    const _Float16* __restrict__ h16, const float* __restrict__ el,
    const float* __restrict__ er, const int* __restrict__ src,
    const float* __restrict__ bias, float* __restrict__ x_attn, int part) {
  const int gw = blockIdx.x * 8 + (threadIdx.x >> 5);
  const int lane = threadIdx.x & 31;
  if (gw >= Nn * Kk) return;
  const int n = gw / Kk, k = gw % Kk;
  float mean_acc = 0.f;
  for (int h = 0; h < 3; ++h) {
    const float erd = er[((size_t)(n * Kk + k)) * 3 + h];
    int   se[DEG];
    float ev[DEG];
    float mx = -INFINITY;
#pragma unroll
    for (int e = 0; e < DEG; ++e) {
      const int s = src[n + e * Nn];
      se[e] = s;
      float x = el[((size_t)(s * Kk + k)) * 3 + h] + erd;
      x = (x > 0.f) ? x : 0.2f * x;  // leaky_relu(.,0.2)
      ev[e] = x;
      mx = fmaxf(mx, x);
    }
    float den = 0.f;
#pragma unroll
    for (int e = 0; e < DEG; ++e) { ev[e] = __expf(ev[e] - mx); den += ev[e]; }
    const float inv = 1.f / (den + 1e-9f);
    float acc = 0.f;
#pragma unroll
    for (int e = 0; e < DEG; ++e)
      acc += ev[e] * inv *
             (float)h16[((size_t)(se[e] * Kk + k)) * 96 + h * 32 + lane];
    mean_acc += acc + bias[h * 32 + lane];
  }
  x_attn[((size_t)(n * Kk + k)) * 96 + part * 32 + lane] = mean_acc * (1.f / 3.f);
}

// softmax over last axis, one block (256 thr) per row
__global__ __launch_bounds__(256) void softmax_rows_kernel(float* __restrict__ p,
                                                           int cols) {
  const int tid = threadIdx.x, lane = tid & 31, wv = tid >> 5;
  float* rp = p + (size_t)blockIdx.x * cols;
  __shared__ float smax[8], ssum[8];
  float m = -INFINITY;
  for (int i = tid; i < cols; i += 256) m = fmaxf(m, rp[i]);
  for (int off = 16; off; off >>= 1) m = fmaxf(m, __shfl_xor(m, off));
  if (lane == 0) smax[wv] = m;
  __syncthreads();
#pragma unroll
  for (int w = 0; w < 8; ++w) m = fmaxf(m, smax[w]);
  float s = 0.f;
  for (int i = tid; i < cols; i += 256) {
    const float e = __expf(rp[i] - m);
    rp[i] = e;
    s += e;
  }
  for (int off = 16; off; off >>= 1) s += __shfl_xor(s, off);
  if (lane == 0) ssum[wv] = s;
  __syncthreads();
  s = 0.f;
#pragma unroll
  for (int w = 0; w < 8; ++w) s += ssum[w];
  const float inv = 1.f / s;
  for (int i = tid; i < cols; i += 256) rp[i] *= inv;
}

// attn_T = softmax(covT @ covT^T, axis=1)  (normalized over rows i, per col j)
__global__ void attn_T_kernel(const float* __restrict__ te,
                              float* __restrict__ attnT) {
  const int j = threadIdx.x, b = blockIdx.x;
  if (j >= Tt) return;
  float lg[Tt];
  float mx = -INFINITY;
  for (int i = 0; i < Tt; ++i) {
    float s = 0.f;
    for (int f = 0; f < Ff; ++f)
      s += te[(((size_t)b * Tt + i) * Nn) * Ff + f] *
           te[(((size_t)b * Tt + j) * Nn) * Ff + f];
    lg[i] = s;
    mx = fmaxf(mx, s);
  }
  float den = 0.f;
  for (int i = 0; i < Tt; ++i) { lg[i] = __expf(lg[i] - mx); den += lg[i]; }
  const float inv = 1.f / den;
  for (int i = 0; i < Tt; ++i) attnT[((size_t)b * Tt + i) * Tt + j] = lg[i] * inv;
}

// temporal GAT: one wave per (kk, i); softmax over j in [0,24), GH=4, GHD=24
__global__ __launch_bounds__(256) void temporal_attn_kernel(
    const _Float16* __restrict__ hg16, const float* __restrict__ elg,
    const float* __restrict__ erg, float* __restrict__ xg) {
  const int gw = blockIdx.x * 8 + (threadIdx.x >> 5);
  const int lane = threadIdx.x & 31;
  if (gw >= Bb * Nn * Tt) return;
  const int kk = gw / Tt, i = gw % Tt;
  for (int h = 0; h < 4; ++h) {
    const float eli = elg[((size_t)(kk * Tt + i)) * 4 + h];
    float ej = -INFINITY;
    if (lane < Tt) {
      float x = eli + erg[((size_t)(kk * Tt + lane)) * 4 + h];
      ej = (x > 0.f) ? x : 0.2f * x;
    }
    float mx = ej;
    for (int off = 16; off; off >>= 1) mx = fmaxf(mx, __shfl_xor(mx, off));
    const float ex = (lane < Tt) ? __expf(ej - mx) : 0.f;
    float s = ex;
    for (int off = 16; off; off >>= 1) s += __shfl_xor(s, off);
    const float alpha = ex / s;
    float acc = 0.f;
    for (int j = 0; j < Tt; ++j) {
      const float aj = __shfl(alpha, j);
      if (lane < Tt)
        acc += aj * (float)hg16[((size_t)(kk * Tt + j)) * 96 + h * 24 + lane];
    }
    if (lane < Tt)
      xg[((size_t)(kk * Tt + i)) * 96 + h * 24 + lane] = acc;
  }
}

// emb = attn_T[kk % B] @ xt ; xg *= sigmoid(emb); temporal = input + xg
__global__ void emb_gate_kernel(const float* __restrict__ attnT,
                                const _Float16* __restrict__ xt16,
                                const float* __restrict__ xg,
                                const float* __restrict__ input,
                                float* __restrict__ temporal,
                                _Float16* __restrict__ concat16) {
  size_t idx = (size_t)blockIdx.x * blockDim.x + threadIdx.x;
  if (idx >= (size_t)Bb * Nn * Tt * Ff) return;
  const int f = idx % Ff;
  const int i = (idx / Ff) % Tt;
  const int kk = idx / ((size_t)Ff * Tt);
  const int b_attn = kk % Bb;  // reference: idx = arange(B*N) % B
  const int b = kk / Nn, n = kk % Nn;
  const float* at = attnT + ((size_t)b_attn * Tt + i) * Tt;
  const _Float16* xp = xt16 + (size_t)kk * Tt * Ff + f;
  float emb = 0.f;
  for (int j = 0; j < Tt; ++j) emb += at[j] * (float)xp[(size_t)j * Ff];
  const float g = 1.f / (1.f + __expf(-emb));
  const float v = xg[idx] * g;
  const size_t row = ((size_t)(b * Tt + i) * Nn + n);  // (b, t=i, n)
  const float tv = input[row * Ff + f] + v;
  temporal[row * Ff + f] = tv;
  concat16[row * 192 + 96 + f] = (_Float16)tv;
}

// spatial = input + x_attn (with the reference's reshape quirk k_read=b*T+t)
__global__ void spatial_assemble_kernel(const float* __restrict__ input,
                                        const float* __restrict__ x_attn,
                                        float* __restrict__ spatial,
                                        _Float16* __restrict__ concat16) {
  size_t idx = (size_t)blockIdx.x * blockDim.x + threadIdx.x;
  if (idx >= (size_t)Bb * Tt * Nn * Ff) return;
  const int f = idx % Ff;
  const int n = (idx / Ff) % Nn;
  const int t = (idx / ((size_t)Ff * Nn)) % Tt;
  const int b = idx / ((size_t)Ff * Nn * Tt);
  const int kread = b * Tt + t;  // (T*B,N,F).reshape(B,T,N,F) reinterpretation
  const float v = input[idx] + x_attn[((size_t)n * Kk + kread) * Ff + f];
  spatial[idx] = v;
  concat16[(idx / Ff) * 192 + f] = (_Float16)v;
}

__global__ void final_fuse_kernel(const float* __restrict__ zlin,
                                  const float* __restrict__ bf,
                                  const float* __restrict__ spatial,
                                  const float* __restrict__ temporal,
                                  const float* __restrict__ input,
                                  float* __restrict__ out) {
  size_t idx = (size_t)blockIdx.x * blockDim.x + threadIdx.x;
  if (idx >= (size_t)Bb * Tt * Nn * Ff) return;
  const int f = idx % Ff;
  const float z = 1.f / (1.f + __expf(-(zlin[idx] + bf[f])));
  out[idx] = z * spatial[idx] + (1.f - z) * temporal[idx] + input[idx];
}

// ---------------------------------------------------------------------------
extern "C" void kernel_launch(void* const* d_in, const int* in_sizes, int n_in,
                              void* d_out, int out_size, void* d_ws,
                              size_t ws_size, hipStream_t stream) {
  (void)in_sizes; (void)n_in; (void)out_size; (void)ws_size;

  const float* input = (const float*)d_in[0];
  const float* spat  = (const float*)d_in[1];
  const float* temb  = (const float*)d_in[2];
  const int*   src_d = (const int*)d_in[3];
  const int*   src_m = (const int*)d_in[5];
  const int*   src_s = (const int*)d_in[7];
  const float* W_d = (const float*)d_in[9],  *al_d = (const float*)d_in[10],
             * ar_d = (const float*)d_in[11], *b_d = (const float*)d_in[12];
  const float* W_m = (const float*)d_in[13], *al_m = (const float*)d_in[14],
             * ar_m = (const float*)d_in[15], *b_m = (const float*)d_in[16];
  const float* W_s = (const float*)d_in[17], *al_s = (const float*)d_in[18],
             * ar_s = (const float*)d_in[19], *b_s = (const float*)d_in[20];
  const float* Wg  = (const float*)d_in[21], *agl = (const float*)d_in[22],
             * agr = (const float*)d_in[23];
  const float* Wf  = (const float*)d_in[24], *bf = (const float*)d_in[25];

  // workspace carve-out (256B aligned)
  char* ws = (char*)d_ws;
  size_t off = 0;
  auto alloc = [&](size_t bytes) -> void* {
    void* p = ws + off;
    off += (bytes + 255) & ~(size_t)255;
    return p;
  };
  const size_t MF = (size_t)Mrows * 96;  // 9,437,184
  _Float16* x16   = (_Float16*)alloc(MF * 2);
  _Float16* xt16  = (_Float16*)alloc(MF * 2);
  _Float16* hd16  = (_Float16*)alloc(MF * 2);
  _Float16* hm16  = (_Float16*)alloc(MF * 2);
  _Float16* hs16  = (_Float16*)alloc(MF * 2);
  _Float16* hg16  = (_Float16*)alloc(MF * 2);
  _Float16* cov16 = (_Float16*)alloc((size_t)Bb * Nn * Ff * 2);
  _Float16* Wd16  = (_Float16*)alloc(96 * 96 * 2);
  _Float16* Wm16  = (_Float16*)alloc(96 * 96 * 2);
  _Float16* Ws16  = (_Float16*)alloc(96 * 96 * 2);
  _Float16* Wg16  = (_Float16*)alloc(96 * 96 * 2);
  _Float16* Wf16  = (_Float16*)alloc(192 * 96 * 2);
  _Float16* concat16 = (_Float16*)alloc((size_t)Mrows * 192 * 2);
  float* el_d = (float*)alloc((size_t)Mrows * 3 * 4);
  float* er_d = (float*)alloc((size_t)Mrows * 3 * 4);
  float* el_m = (float*)alloc((size_t)Mrows * 3 * 4);
  float* er_m = (float*)alloc((size_t)Mrows * 3 * 4);
  float* el_s = (float*)alloc((size_t)Mrows * 3 * 4);
  float* er_s = (float*)alloc((size_t)Mrows * 3 * 4);
  float* elg  = (float*)alloc((size_t)Mrows * 4 * 4);
  float* erg  = (float*)alloc((size_t)Mrows * 4 * 4);
  float* x_attn   = (float*)alloc(MF * 4);
  float* xg       = (float*)alloc(MF * 4);
  float* spatial  = (float*)alloc(MF * 4);
  float* temporal = (float*)alloc(MF * 4);
  float* zlin     = (float*)alloc(MF * 4);
  float* attnT    = (float*)alloc((size_t)Bb * Tt * Tt * 4);

  float* out_main = (float*)d_out;              // (B,T,N,F)
  float* out_attnS = out_main + MF;             // (B,N,N)

  const int EW = 9437184;  // B*T*N*F elements
  const dim3 blk256(256);

  // 1. stage x = input+spatial (N,K,F) and xt = input (B*N,T,F) as f16
  prep_kernel<<<(EW + 255) / 256, blk256, 0, stream>>>(input, spat, x16, xt16);

  // 2. weight / cov conversions to f16
  cvt_f16_kernel<<<(9216 + 255) / 256, blk256, 0, stream>>>(W_d, Wd16, 9216);
  cvt_f16_kernel<<<(9216 + 255) / 256, blk256, 0, stream>>>(W_m, Wm16, 9216);
  cvt_f16_kernel<<<(9216 + 255) / 256, blk256, 0, stream>>>(W_s, Ws16, 9216);
  cvt_f16_kernel<<<(9216 + 255) / 256, blk256, 0, stream>>>(Wg, Wg16, 9216);
  cvt_f16_kernel<<<(18432 + 255) / 256, blk256, 0, stream>>>(Wf, Wf16, 18432);
  cov0_cvt_kernel<<<(Bb * Nn * Ff + 255) / 256, blk256, 0, stream>>>(spat, cov16);

  // 3. WMMA projection GEMMs: h_{d,m,s} = x@W, hg = xt@Wg  (M=98304, KD=96)
  wmma_gemm_rm<96><<<Mrows / 16, 192, 0, stream>>>(x16, Wd16, hd16, nullptr);
  wmma_gemm_rm<96><<<Mrows / 16, 192, 0, stream>>>(x16, Wm16, hm16, nullptr);
  wmma_gemm_rm<96><<<Mrows / 16, 192, 0, stream>>>(x16, Ws16, hs16, nullptr);
  wmma_gemm_rm<96><<<Mrows / 16, 192, 0, stream>>>(xt16, Wg16, hg16, nullptr);

  // 4. attention coefficient reductions
  el_er_kernel<<<(Mrows * 3 + 255) / 256, blk256, 0, stream>>>(hd16, al_d, ar_d, el_d, er_d, Mrows, 3, 32);
  el_er_kernel<<<(Mrows * 3 + 255) / 256, blk256, 0, stream>>>(hm16, al_m, ar_m, el_m, er_m, Mrows, 3, 32);
  el_er_kernel<<<(Mrows * 3 + 255) / 256, blk256, 0, stream>>>(hs16, al_s, ar_s, el_s, er_s, Mrows, 3, 32);
  el_er_kernel<<<(Mrows * 4 + 255) / 256, blk256, 0, stream>>>(hg16, agl, agr, elg, erg, Mrows, 4, 24);

  // 5. spatial GAT aggregation (8 fixed edges per node; wave per (n,k))
  const int aggBlocks = (Nn * Kk) / 8;
  gat_aggregate_kernel<<<aggBlocks, blk256, 0, stream>>>(hd16, el_d, er_d, src_d, b_d, x_attn, 0);
  gat_aggregate_kernel<<<aggBlocks, blk256, 0, stream>>>(hm16, el_m, er_m, src_m, b_m, x_attn, 1);
  gat_aggregate_kernel<<<aggBlocks, blk256, 0, stream>>>(hs16, el_s, er_s, src_s, b_s, x_attn, 2);

  // 6. attn_S = softmax(cov0 @ cov0^T) via WMMA + row softmax (written to d_out)
  wmma_gemm_nt_covS<<<dim3(Nn / 16, Nn / 128, Bb), 256, 0, stream>>>(cov16, out_attnS);
  softmax_rows_kernel<<<Bb * Nn, blk256, 0, stream>>>(out_attnS, Nn);

  // 7. temporal branch
  attn_T_kernel<<<Bb, 32, 0, stream>>>(temb, attnT);
  temporal_attn_kernel<<<(Bb * Nn * Tt) / 8, blk256, 0, stream>>>(hg16, elg, erg, xg);
  emb_gate_kernel<<<(EW + 255) / 256, blk256, 0, stream>>>(attnT, xt16, xg, input, temporal, concat16);

  // 8. spatial residual + concat staging
  spatial_assemble_kernel<<<(EW + 255) / 256, blk256, 0, stream>>>(input, x_attn, spatial, concat16);

  // 9. fusion GEMM (M=98304, KD=192) + gated residual output
  wmma_gemm_rm<192><<<Mrows / 16, 192, 0, stream>>>(concat16, Wf16, nullptr, zlin);
  final_fuse_kernel<<<(EW + 255) / 256, blk256, 0, stream>>>(zlin, bf, spatial, temporal, input, out_main);
}
